// Attention_4612794876520
// MI455X (gfx1250) — matrix operand order repaired and checked by reading, not compiled, not run on hardware
//
#include <hip/hip_runtime.h>
#include <hip/hip_bf16.h>

// ---------------------------------------------------------------------------
// GQA attention (B=2,T=2048,E=2048,H=16,D=128,KVH=8) with RoPE, causal softmax.
// Matmuls via v_wmma_f32_16x16x32_bf16; row-major tile staging via the
// Tensor Data Mover (tensor_load_to_lds + s_wait_tensorcnt) on gfx1250.
// ---------------------------------------------------------------------------

#define BB   2
#define TT   2048
#define EE   2048
#define HH   16
#define DD   128
#define KVH  8

typedef __attribute__((ext_vector_type(16))) __bf16 v16bf;
typedef __attribute__((ext_vector_type(8)))  float  v8f;

union FragU {                 // 32-byte bf16 fragment <-> 2x uint4
    uint4 q[2];
    v16bf v;
};
union U4U {                   // uint4 <-> 8 ushorts
    uint4 q;
    unsigned short u[8];
};

__device__ __forceinline__ unsigned short f2bf(float f) {
    union { float f; unsigned u; } x; x.f = f;
    unsigned r = x.u + 0x7FFFu + ((x.u >> 16) & 1u);   // RNE
    return (unsigned short)(r >> 16);
}
__device__ __forceinline__ float bf2f(unsigned short h) {
    union { unsigned u; float f; } x; x.u = ((unsigned)h) << 16;
    return x.f;
}

// ---------------------------------------------------------------------------
// Tensor Data Mover: 2D bf16 tile -> LDS with row padding (ISA ch. 8 D#).
// ---------------------------------------------------------------------------
#if __has_builtin(__builtin_amdgcn_tensor_load_to_lds)
#define HAS_TDM 1
typedef unsigned int v4u __attribute__((ext_vector_type(4)));
typedef int          v8i __attribute__((ext_vector_type(8)));
typedef int          v4i __attribute__((ext_vector_type(4)));

// d0 = tile width (bf16 elems, row-major contiguous), d1 = tile rows,
// stride0 = global row stride (elems).  pad codes per ISA: interval 0=>2 DW,
// 1=>4, 2=>8, 3=>16, 5=>64; amount code c => (c+1) DWORDs.
__device__ __forceinline__ void tdm_load_2d(const unsigned short* gaddr,
                                            unsigned lds_off,
                                            unsigned d0, unsigned d1,
                                            unsigned long long stride0,
                                            unsigned pad_int_code,
                                            unsigned pad_amt_code) {
    unsigned long long ga = (unsigned long long)(size_t)gaddr;
    v4u g0;
    g0.x = 1u;                                          // count=1 (valid), user
    g0.y = lds_off;                                     // lds_addr (bytes)
    g0.z = (unsigned)(ga & 0xFFFFFFFFull);              // global_addr[31:0]
    g0.w = (unsigned)((ga >> 32) & 0x01FFFFFFull)       // global_addr[56:32]
         | (2u << 30);                                  // type=2 (image)
    v8i g1;
    g1[0] = (int)((1u << 16)                            // data_size = 2 bytes
                | (1u << 20)                            // pad_enable
                | (pad_int_code << 22)
                | (pad_amt_code << 25));
    g1[1] = (int)((d0 & 0xFFFFu) << 16);                // tensor_dim0[15:0]
    g1[2] = (int)((d0 >> 16) | ((d1 & 0xFFFFu) << 16)); // dim0[31:16]|dim1[15:0]
    g1[3] = (int)((d1 >> 16) | (d0 << 16));             // dim1[31:16]|tile_dim0
    g1[4] = (int)(d1 & 0xFFFFu);                        // tile_dim1 (tile_dim2=0)
    g1[5] = (int)(stride0 & 0xFFFFFFFFull);             // dim0_stride[31:0]
    g1[6] = (int)((stride0 >> 32) & 0xFFFFull);         // dim0_stride[47:32]
    g1[7] = 0;
    v4i z4 = {0, 0, 0, 0};
#if defined(__clang_major__) && __clang_major__ >= 23
    v8i z8 = {0, 0, 0, 0, 0, 0, 0, 0};
    __builtin_amdgcn_tensor_load_to_lds(g0, g1, z4, z4, z8, 0);
#else
    __builtin_amdgcn_tensor_load_to_lds(g0, g1, z4, z4, 0);
#endif
}
__device__ __forceinline__ void tdm_wait() {
    __builtin_amdgcn_s_wait_tensorcnt(0);
}
__device__ __forceinline__ unsigned lds_off_of(const void* p) {
    return (unsigned)(size_t)p;                         // LDS flat addr[31:0]
}
#else
#define HAS_TDM 0
#endif

// ---------------------------------------------------------------------------
// f32 -> bf16 cast
// ---------------------------------------------------------------------------
__global__ void cast_f32_bf16(const float* __restrict__ in,
                              unsigned short* __restrict__ out, int n) {
    int i = blockIdx.x * blockDim.x + threadIdx.x;
    if (i < n) out[i] = f2bf(in[i]);
}

// ---------------------------------------------------------------------------
// bf16 GEMM: C(MxN) = A(MxK) * B(KxN).  Block = 256 threads = 8 waves.
// Workgroup tile 128x128; wave grid 4x2; each wave 32x64 (2x4 wmma accums).
// A tile staged by TDM (row-major + pad); B tile staged transposed manually.
// ---------------------------------------------------------------------------
template <bool OUT_F32>
__global__ __launch_bounds__(256)
void gemm_bf16(const unsigned short* __restrict__ A,
               const unsigned short* __restrict__ Bm,
               unsigned short* __restrict__ Cbf,
               float* __restrict__ Cf,
               int M, int N, int K) {
    constexpr int LDT = 48;                       // padded LDS stride (96B rows)
    __shared__ unsigned short As [128 * LDT];     // A tile  128 x 32
    __shared__ unsigned short BsT[128 * LDT];     // B tile transposed: 128(N) x 32(K)

    const int tid  = threadIdx.x;
    const int lane = tid & 31;
    const int wave = tid >> 5;
    const int wr   = wave & 3;                    // wave row   (4 x 32 rows)
    const int wc   = wave >> 2;                   // wave col   (2 x 64 cols)
    const int rb   = blockIdx.y * 128;
    const int cb   = blockIdx.x * 128;

    v8f acc[2][4];
    #pragma unroll
    for (int m = 0; m < 2; ++m)
        #pragma unroll
        for (int n = 0; n < 4; ++n) acc[m][n] = (v8f){};

    for (int k0 = 0; k0 < K; k0 += 32) {
        // --- stage A tile (128 rows x 32 K) ---
#if HAS_TDM
        if (wave == 0) {
            // tile row 32 elems = 64B = 16 DW (code 3); pad 16 elems = 8 DW (code 7)
            tdm_load_2d(A + (size_t)rb * K + k0, lds_off_of(&As[0]),
                        32u, 128u, (unsigned long long)K, 3u, 7u);
        }
#else
        {
            int row = tid >> 1, seg = tid & 1;
            const uint4* g =
                reinterpret_cast<const uint4*>(A + (size_t)(rb + row) * K + k0 + seg * 16);
            uint4 a0 = g[0], a1 = g[1];
            uint4* s = reinterpret_cast<uint4*>(&As[row * LDT + seg * 16]);
            s[0] = a0; s[1] = a1;
        }
#endif
        // --- stage B tile transposed (cols -> rows) ---
        {
            int krow = tid >> 3, seg = tid & 7;
            const uint4* g =
                reinterpret_cast<const uint4*>(Bm + (size_t)(k0 + krow) * N + cb + seg * 16);
            U4U b0, b1; b0.q = g[0]; b1.q = g[1];
            #pragma unroll
            for (int e = 0; e < 8; ++e) {
                BsT[(seg * 16 + e    ) * LDT + krow] = b0.u[e];
                BsT[(seg * 16 + e + 8) * LDT + krow] = b1.u[e];
            }
        }
#if HAS_TDM
        if (wave == 0) tdm_wait();
#endif
        __syncthreads();

        #pragma unroll
        for (int m = 0; m < 2; ++m) {
            FragU a;
            int arow = wr * 32 + m * 16 + (lane & 15);
            int kb2  = (lane >> 4) * 16;                      // A: 16 contiguous K
            a.q[0] = *reinterpret_cast<const uint4*>(&As[arow * LDT + kb2]);
            a.q[1] = *reinterpret_cast<const uint4*>(&As[arow * LDT + kb2 + 8]);
            #pragma unroll
            for (int n = 0; n < 4; ++n) {
                FragU b;
                int bcol = wc * 64 + n * 16 + (lane & 15);
                int kbb  = (lane >> 4) * 16;                  // B: 16 contiguous K
                const uint4* pb =
                    reinterpret_cast<const uint4*>(&BsT[bcol * LDT + kbb]);
                b.q[0] = pb[0]; b.q[1] = pb[1];
                acc[m][n] = __builtin_amdgcn_wmma_f32_16x16x32_bf16(
                    false, a.v, false, b.v, (short)0, acc[m][n], false, false);
            }
        }
        __syncthreads();
    }

    // --- epilogue: C layout M = j + 8*(lane>>4), N = lane&15 ---
    #pragma unroll
    for (int m = 0; m < 2; ++m)
        #pragma unroll
        for (int n = 0; n < 4; ++n)
            #pragma unroll
            for (int j = 0; j < 8; ++j) {
                int row = rb + wr * 32 + m * 16 + 8 * (lane >> 4) + j;
                int col = cb + wc * 64 + n * 16 + (lane & 15);
                float v = acc[m][n][j];
                if (OUT_F32) Cf [(size_t)row * N + col] = v;
                else         Cbf[(size_t)row * N + col] = f2bf(v);
            }
}

// ---------------------------------------------------------------------------
// RoPE (in place on bf16 (B,T,heads,D)); one thread per complex pair.
// ---------------------------------------------------------------------------
__global__ void rope_bf16(unsigned short* __restrict__ buf, int heads, int npairs) {
    int i = blockIdx.x * blockDim.x + threadIdx.x;
    if (i >= npairs) return;
    const int HP = DD / 2;                                   // 64 pairs per head
    int p   = i % HP;
    int tmp = i / HP;
    int hh  = tmp % heads;
    int bt  = tmp / heads;
    int t   = bt % TT;
    // inv_freq = theta^(-2p/D); ln(10000)=9.210340371976184
    float angle = (float)t * __expf(-((float)p / 64.0f) * 9.210340371976184f);
    float c = __cosf(angle), s = __sinf(angle);
    unsigned short* ptr = buf + (size_t)bt * heads * DD + hh * DD + 2 * p;
    float a = bf2f(ptr[0]), b = bf2f(ptr[1]);
    ptr[0] = f2bf(a * c - b * s);
    ptr[1] = f2bf(a * s + b * c);
}

// ---------------------------------------------------------------------------
// Flash attention: grid (T/64, H, B); block 128 = 4 waves; wave owns 16 q rows.
// K tile staged by TDM; V tile staged transposed manually.
// S = Q*K^T and O += P*V both via v_wmma_f32_16x16x32_bf16.
// ---------------------------------------------------------------------------
__global__ __launch_bounds__(128)
void flash_attn(const unsigned short* __restrict__ qbuf,   // (B*T, H*D) bf16
                const unsigned short* __restrict__ kbuf,   // (B*T, KVH*D)
                const unsigned short* __restrict__ vbuf,   // (B*T, KVH*D)
                unsigned short* __restrict__ obuf) {       // (B*T, H*D)
    constexpr int LKT = 136;   // K tile stride  (272B rows)
    constexpr int LVT = 72;    // V^T tile stride (144B rows)
    __shared__ unsigned short Kt [64 * LKT];        // 64 kv rows x 128 d
    __shared__ unsigned short VtT[128 * LVT];       // 128 d x 64 kv
    __shared__ unsigned short Pt [4][16 * LVT];     // per-wave P tile 16 x 64

    const int tid  = threadIdx.x;
    const int lane = tid & 31;
    const int wave = tid >> 5;
    const int qb   = blockIdx.x;
    const int h    = blockIdx.y;
    const int b    = blockIdx.z;
    const int kvh  = h >> 1;                        // GQA: 2 q heads per kv head
    const int qlo  = qb * 64 + wave * 16;

    // ---- load Q fragments (16 rows x 128 d) once ----
    v16bf qf[4];
    {
        int qrow = qlo + (lane & 15);
        const unsigned short* qp =
            qbuf + ((size_t)(b * TT + qrow)) * (HH * DD) + h * DD;
        int kb2 = (lane >> 4) * 16;
        #pragma unroll
        for (int c = 0; c < 4; ++c) {
            FragU a;
            a.q[0] = *reinterpret_cast<const uint4*>(qp + c * 32 + kb2);
            a.q[1] = *reinterpret_cast<const uint4*>(qp + c * 32 + kb2 + 8);
            qf[c] = a.v;
        }
    }

    v8f o[8];
    #pragma unroll
    for (int d = 0; d < 8; ++d) o[d] = (v8f){};
    float rmax[8], rsum[8];
    #pragma unroll
    for (int j = 0; j < 8; ++j) { rmax[j] = -3.0e38f; rsum[j] = 0.0f; }

    const int nkb = qb + 1;                         // causal: kv blocks 0..qb
    for (int kb = 0; kb < nkb; ++kb) {
        __syncthreads();
        // ---- stage K (row major, TDM) ----
#if HAS_TDM
        if (wave == 0) {
            // tile row 128 elems = 256B = 64 DW (code 5); pad 16B = 4 DW (code 3)
            tdm_load_2d(kbuf + ((size_t)(b * TT + kb * 64)) * (KVH * DD) + kvh * DD,
                        lds_off_of(&Kt[0]), 128u, 64u,
                        (unsigned long long)(KVH * DD), 5u, 3u);
        }
#else
        {
            int r    = tid >> 1;
            int half = (tid & 1) * 64;
            const uint4* gk = reinterpret_cast<const uint4*>(
                kbuf + ((size_t)(b * TT + kb * 64 + r)) * (KVH * DD) + kvh * DD + half);
            uint4* sk = reinterpret_cast<uint4*>(&Kt[r * LKT + half]);
            #pragma unroll
            for (int i = 0; i < 8; ++i) sk[i] = gk[i];
        }
#endif
        // ---- stage V transposed (all threads) ----
        {
            int r    = tid >> 1;
            int half = (tid & 1) * 64;
            const uint4* gv = reinterpret_cast<const uint4*>(
                vbuf + ((size_t)(b * TT + kb * 64 + r)) * (KVH * DD) + kvh * DD + half);
            #pragma unroll
            for (int i = 0; i < 8; ++i) {
                U4U t; t.q = gv[i];
                #pragma unroll
                for (int e = 0; e < 8; ++e)
                    VtT[(half + i * 8 + e) * LVT + r] = t.u[e];
            }
        }
#if HAS_TDM
        if (wave == 0) tdm_wait();
#endif
        __syncthreads();

        // ---- S = Q * K^T (4 tiles of 16x16) ----
        v8f  s[4];
        bool valid[4];
        #pragma unroll
        for (int nt = 0; nt < 4; ++nt) {
            int kvb = kb * 64 + nt * 16;
            valid[nt] = (kvb <= qlo + 15);
            if (!valid[nt]) continue;
            v8f acc = (v8f){};
            int krow = nt * 16 + (lane & 15);
            int kbb  = (lane >> 4) * 16;
            #pragma unroll
            for (int c = 0; c < 4; ++c) {
                FragU bfr;
                const uint4* pb =
                    reinterpret_cast<const uint4*>(&Kt[krow * LKT + c * 32 + kbb]);
                bfr.q[0] = pb[0]; bfr.q[1] = pb[1];
                acc = __builtin_amdgcn_wmma_f32_16x16x32_bf16(
                    false, qf[c], false, bfr.v, (short)0, acc, false, false);
            }
            // scale + causal mask
            #pragma unroll
            for (int j = 0; j < 8; ++j) {
                int qrow = qlo + 8 * (lane >> 4) + j;
                int kv   = kvb + (lane & 15);
                float sv = acc[j] * 0.08838834764831845f;   // 1/sqrt(128)
                if (kv > qrow) sv = -3.0e38f;
                acc[j] = sv;
            }
            s[nt] = acc;
        }

        // ---- online softmax (row stats via 16-lane xor reduction) ----
        #pragma unroll
        for (int j = 0; j < 8; ++j) {
            float m = -3.0e38f;
            #pragma unroll
            for (int nt = 0; nt < 4; ++nt)
                if (valid[nt]) m = fmaxf(m, s[nt][j]);
            #pragma unroll
            for (int msk = 1; msk < 16; msk <<= 1)
                m = fmaxf(m, __shfl_xor(m, msk, 32));
            float nm = fmaxf(rmax[j], m);
            float so = __expf(rmax[j] - nm);
            rmax[j]  = nm;
            float rs = 0.0f;
            #pragma unroll
            for (int nt = 0; nt < 4; ++nt) {
                if (valid[nt]) {
                    float p = __expf(s[nt][j] - nm);
                    s[nt][j] = p;
                    rs += p;
                }
            }
            #pragma unroll
            for (int msk = 1; msk < 16; msk <<= 1)
                rs += __shfl_xor(rs, msk, 32);
            rsum[j] = rsum[j] * so + rs;
            #pragma unroll
            for (int d = 0; d < 8; ++d) o[d][j] *= so;
            // stash P (C layout -> LDS row major), zero masked tiles
            int prow = 8 * (lane >> 4) + j;
            #pragma unroll
            for (int nt = 0; nt < 4; ++nt) {
                unsigned short pv = valid[nt] ? f2bf(s[nt][j]) : (unsigned short)0;
                Pt[wave][prow * LVT + nt * 16 + (lane & 15)] = pv;
            }
        }

        // ---- O += P * V (contract 64 kv in two 32-chunks) ----
        #pragma unroll
        for (int c = 0; c < 2; ++c) {
            FragU pa;
            int prr = lane & 15;
            int kb2 = (lane >> 4) * 16;
            const unsigned short* pbase = &Pt[wave][prr * LVT + c * 32 + kb2];
            pa.q[0] = *reinterpret_cast<const uint4*>(pbase);
            pa.q[1] = *reinterpret_cast<const uint4*>(pbase + 8);
            #pragma unroll
            for (int dt = 0; dt < 8; ++dt) {
                FragU vb;
                int dcol = dt * 16 + (lane & 15);
                int kvb2 = (lane >> 4) * 16;
                const uint4* pv =
                    reinterpret_cast<const uint4*>(&VtT[dcol * LVT + c * 32 + kvb2]);
                vb.q[0] = pv[0]; vb.q[1] = pv[1];
                o[dt] = __builtin_amdgcn_wmma_f32_16x16x32_bf16(
                    false, pa.v, false, vb.v, (short)0, o[dt], false, false);
            }
        }
    }

    // ---- normalize + write O (bf16) ----
    #pragma unroll
    for (int j = 0; j < 8; ++j) {
        float inv = 1.0f / rsum[j];
        int q = qlo + 8 * (lane >> 4) + j;
        unsigned short* op = obuf + ((size_t)(b * TT + q)) * (HH * DD) + h * DD;
        #pragma unroll
        for (int dt = 0; dt < 8; ++dt)
            op[dt * 16 + (lane & 15)] = f2bf(o[dt][j] * inv);
    }
}

// ---------------------------------------------------------------------------
// Host launcher
// ---------------------------------------------------------------------------
extern "C" void kernel_launch(void* const* d_in, const int* in_sizes, int n_in,
                              void* d_out, int out_size, void* d_ws, size_t ws_size,
                              hipStream_t stream) {
    const float* x  = (const float*)d_in[0];
    /* d_in[1] = mask (plain causal tril; positions used instead) */
    const float* wq = (const float*)d_in[2];
    const float* wk = (const float*)d_in[3];
    const float* wv = (const float*)d_in[4];
    const float* wo = (const float*)d_in[5];
    float* out = (float*)d_out;

    const int M = BB * TT;                       // 4096 rows
    char* base = (char*)d_ws;
    size_t off = 0;
    auto alloc = [&](size_t bytes) {
        size_t o = off;
        off = (off + bytes + 255) & ~(size_t)255;
        return (unsigned short*)(base + o);
    };
    unsigned short* xb  = alloc((size_t)M * EE * 2);
    unsigned short* wqb = alloc((size_t)EE * (HH * DD) * 2);
    unsigned short* wkb = alloc((size_t)EE * (KVH * DD) * 2);
    unsigned short* wvb = alloc((size_t)EE * (KVH * DD) * 2);
    unsigned short* wob = alloc((size_t)(HH * DD) * EE * 2);
    unsigned short* qbb = alloc((size_t)M * (HH * DD) * 2);
    unsigned short* kbb = alloc((size_t)M * (KVH * DD) * 2);
    unsigned short* vbb = alloc((size_t)M * (KVH * DD) * 2);
    unsigned short* obb = alloc((size_t)M * (HH * DD) * 2);
    (void)ws_size;

    auto cast = [&](const float* src, unsigned short* dst, size_t n) {
        cast_f32_bf16<<<(unsigned)((n + 255) / 256), 256, 0, stream>>>(src, dst, (int)n);
    };
    cast(x,  xb,  (size_t)M * EE);
    cast(wq, wqb, (size_t)EE * HH * DD);
    cast(wk, wkb, (size_t)EE * KVH * DD);
    cast(wv, wvb, (size_t)EE * KVH * DD);
    cast(wo, wob, (size_t)HH * DD * EE);

    // QKV projections (bf16 out)
    gemm_bf16<false><<<dim3((HH * DD) / 128, M / 128), 256, 0, stream>>>(
        xb, wqb, qbb, nullptr, M, HH * DD, EE);
    gemm_bf16<false><<<dim3((KVH * DD) / 128, M / 128), 256, 0, stream>>>(
        xb, wkb, kbb, nullptr, M, KVH * DD, EE);
    gemm_bf16<false><<<dim3((KVH * DD) / 128, M / 128), 256, 0, stream>>>(
        xb, wvb, vbb, nullptr, M, KVH * DD, EE);

    // RoPE on Q and K
    {
        int npq = M * HH * (DD / 2);
        int npk = M * KVH * (DD / 2);
        rope_bf16<<<(npq + 255) / 256, 256, 0, stream>>>(qbb, HH, npq);
        rope_bf16<<<(npk + 255) / 256, 256, 0, stream>>>(kbb, KVH, npk);
    }

    // Flash attention
    flash_attn<<<dim3(TT / 64, HH, BB), 128, 0, stream>>>(qbb, kbb, vbb, obb);

    // Output projection (f32 out)
    gemm_bf16<true><<<dim3(EE / 128, M / 128), 256, 0, stream>>>(
        obb, wob, nullptr, out, M, EE, HH * DD);
    (void)in_sizes; (void)n_in; (void)out_size;
}
